// ClusGCN_19387482374703
// MI455X (gfx1250) — matrix operand order repaired and checked by reading, not compiled, not run on hardware
//
#include <hip/hip_runtime.h>

typedef __attribute__((ext_vector_type(16))) _Float16 v16h;
typedef __attribute__((ext_vector_type(8)))  float    v8f;

#define NUM_USERS_C 100000
#define NUM_ITEMS_C 50000
#define NUM_NODES_C 150000
#define EMB 128
#define NNZ_TOT 2000000
#define NUM_CLUSTERS_C 10

// ---------------- quantile via radix-select (MSB-first, 4x8-bit passes) ----------------

__global__ void select_init_kernel(unsigned* state, int rank) {
    if (threadIdx.x == 0) { state[0] = 0u; state[1] = (unsigned)rank; }
}

__global__ void hist_kernel(const float* __restrict__ v, unsigned* __restrict__ hist,
                            const unsigned* __restrict__ state, int pass, int n) {
    unsigned prefix = state[0];
    for (int i = blockIdx.x * blockDim.x + threadIdx.x; i < n; i += gridDim.x * blockDim.x) {
        unsigned key = __float_as_uint(v[i]);   // all values positive -> bit pattern is monotone
        bool ok = (pass == 0) || ((key >> (32 - 8 * pass)) == prefix);
        if (ok) atomicAdd(&hist[(key >> (24 - 8 * pass)) & 255u], 1u);
    }
}

__global__ void pick_kernel(unsigned* __restrict__ hist, unsigned* __restrict__ state,
                            int pass, unsigned* __restrict__ result, int which) {
    __shared__ unsigned h[256];
    h[threadIdx.x] = hist[threadIdx.x];
    hist[threadIdx.x] = 0u;                    // reset for next pass
    __syncthreads();
    if (threadIdx.x == 0) {
        unsigned k = state[1];
        unsigned b = 0;
        for (; b < 255u; ++b) {
            if (k < h[b]) break;
            k -= h[b];
        }
        state[0] = (state[0] << 8) | b;
        state[1] = k;
        if (pass == 3) result[which] = state[0];
    }
}

__global__ void deg_kernel(const float* __restrict__ ud, const unsigned* __restrict__ qbits,
                           float* __restrict__ deg, int n) {
    float s0 = __uint_as_float(qbits[0]);
    float s1 = __uint_as_float(qbits[1]);
    float q  = 0.75f * s0 + 0.25f * s1;        // jnp.quantile linear interp at 74999.25
    int i = blockIdx.x * blockDim.x + threadIdx.x;
    if (i < n) deg[i] = fminf(1.0f, ud[i] / q);
}

// ---------------- dense feature transform via WMMA f16->f32 ----------------
// out[m][n] = emb[m][n] + sum_K feat[m][K]*W[n][K] + bias(n), K padded 28(or 19)->32

__device__ __forceinline__ float feat_at(const float* fA, int fa, const float* fB, int fb,
                                         int u, int K) {
    if (K < fa) return fA[(size_t)u * fa + K];
    K -= fa;
    if (K < fb) return fB[(size_t)u * fb + K];
    return 0.0f;
}
__device__ __forceinline__ float w_at(const float* W1, int fa, const float* W2, int fb,
                                      int n, int K) {
    if (K < fa) return W1[(size_t)n * fa + K];
    K -= fa;
    if (K < fb) return W2[(size_t)n * fb + K];
    return 0.0f;
}

__global__ void feat_wmma_kernel(const float* __restrict__ fA, int fa,
                                 const float* __restrict__ fB, int fb,
                                 const float* __restrict__ W1, const float* __restrict__ W2,
                                 const float* __restrict__ b1, const float* __restrict__ b2,
                                 const float* __restrict__ emb,
                                 float* __restrict__ out) {
    const int lane  = threadIdx.x & 31;
    const int wave  = threadIdx.x >> 5;     // 8 waves -> 8 N-tiles of 16 (covers EMB=128)
    const int hi    = lane >> 4;            // half-wave selector
    const int lo    = lane & 15;
    const int rbase = blockIdx.x * 16;      // 16 rows per block
    const int nbase = wave * 16;

    // A (16x32 f16): row M=lo; element e -> K = e + (e>=8?8:0) + hi*8
    v16h a;
#pragma unroll
    for (int e = 0; e < 16; ++e) {
        int K = e + ((e >= 8) ? 8 : 0) + hi * 8;
        a[e] = (_Float16)feat_at(fA, fa, fB, fb, rbase + lo, K);
    }
    // B (32x16 f16): col N=lo; element e -> K = e + (e>=8?8:0) + hi*8
    v16h b;
#pragma unroll
    for (int e = 0; e < 16; ++e) {
        int K = e + ((e >= 8) ? 8 : 0) + hi * 8;
        b[e] = (_Float16)w_at(W1, fa, W2, fb, nbase + lo, K);
    }
    v8f c = {};
    c = __builtin_amdgcn_wmma_f32_16x16x32_f16(false, a, false, b, (short)0, c, false, false);

    const int n = nbase + lo;
    const float bias = b1[n] + (b2 ? b2[n] : 0.0f);
#pragma unroll
    for (int r = 0; r < 8; ++r) {
        int m = rbase + r + hi * 8;         // C/D: VGPR r -> row r + hi*8, col lo
        out[(size_t)m * EMB + n] = c[r] + emb[(size_t)m * EMB + n] + bias;
    }
}

// ---------------- cluster statistics + fused blend ----------------

__global__ void cluster_kernel(const float* __restrict__ u0, const float* __restrict__ deg,
                               const int* __restrict__ cl, float* __restrict__ gsum,
                               float* __restrict__ gcnt) {
    __shared__ float s_sum[NUM_CLUSTERS_C * EMB];
    __shared__ float s_cnt[NUM_CLUSTERS_C];
    for (int j = threadIdx.x; j < NUM_CLUSTERS_C * EMB; j += blockDim.x) s_sum[j] = 0.0f;
    if (threadIdx.x < NUM_CLUSTERS_C) s_cnt[threadIdx.x] = 0.0f;
    __syncthreads();
    const int d = threadIdx.x;              // blockDim.x == 128: one thread per dim
    int ub = blockIdx.x * 512;
    int ue = ub + 512; if (ue > NUM_USERS_C) ue = NUM_USERS_C;
    for (int u = ub; u < ue; ++u) {
        int   c  = cl[u];
        float dg = deg[u];
        s_sum[c * EMB + d] += dg * u0[(size_t)u * EMB + d];  // race-free: thread owns dim d
        if (d == 0) s_cnt[c] += dg;
    }
    __syncthreads();
    for (int j = threadIdx.x; j < NUM_CLUSTERS_C * EMB; j += blockDim.x)
        unsafeAtomicAdd(&gsum[j], s_sum[j]);
    if (threadIdx.x < NUM_CLUSTERS_C) unsafeAtomicAdd(&gcnt[threadIdx.x], s_cnt[threadIdx.x]);
}

__global__ void fused_kernel(float* __restrict__ x, const float* __restrict__ deg,
                             const int* __restrict__ cl, const float* __restrict__ gsum,
                             const float* __restrict__ gcnt) {
    size_t i = (size_t)blockIdx.x * blockDim.x + threadIdx.x;
    if (i >= (size_t)NUM_USERS_C * EMB) return;
    int u = (int)(i / EMB), d = (int)(i % EMB);
    int c = cl[u];
    float a    = deg[u];
    float mean = gsum[c * EMB + d] / gcnt[c];
    x[i] = a * x[i] + (1.0f - a) * mean;
}

// ---------------- SpMM: one wave per edge, 8 edges per block, b128 gathers, fp32 hw atomics ----------------
// Dense operand (150k x 128 f32 = 76.8 MB) fits in the 192 MB L2 -> gathers and atomic
// scatters resolve on-chip; HBM only sees the streaming passes.

__global__ void spmm_kernel(const float* __restrict__ vals, const int* __restrict__ row,
                            const int* __restrict__ col, const float* __restrict__ x,
                            float* __restrict__ y, int nnz) {
    int e = blockIdx.x * blockDim.y + threadIdx.y;
    if (e >= nnz) return;
    int lane = threadIdx.x;                 // 0..31, 4 dims per lane
    float v  = vals[e];
    size_t r = (size_t)row[e] * EMB;
    size_t c = (size_t)col[e] * EMB;
    const float4* xv = (const float4*)(x + c);
    float4 xx = xv[lane];                   // global_load_b128, wave fetches full 512B row
    float* yp = y + r + lane * 4;
    unsafeAtomicAdd(yp + 0, v * xx.x);
    unsafeAtomicAdd(yp + 1, v * xx.y);
    unsafeAtomicAdd(yp + 2, v * xx.z);
    unsafeAtomicAdd(yp + 3, v * xx.w);
}

// ---------------- streaming helpers ----------------

__global__ void copy_kernel(float* __restrict__ o, const float* __restrict__ a, size_t n) {
    size_t i = (size_t)blockIdx.x * blockDim.x + threadIdx.x;
    if (i < n) o[i] = a[i];
}
__global__ void add_kernel(float* __restrict__ o, const float* __restrict__ a, size_t n) {
    size_t i = (size_t)blockIdx.x * blockDim.x + threadIdx.x;
    if (i < n) o[i] += a[i];
}
__global__ void add_scale_kernel(float* __restrict__ o, const float* __restrict__ a,
                                 float s, size_t n) {
    size_t i = (size_t)blockIdx.x * blockDim.x + threadIdx.x;
    if (i < n) o[i] = (o[i] + a[i]) * s;
}

// ---------------- driver ----------------

extern "C" void kernel_launch(void* const* d_in, const int* in_sizes, int n_in,
                              void* d_out, int out_size, void* d_ws, size_t ws_size,
                              hipStream_t stream) {
    const float* user_emb    = (const float*)d_in[0];
    const float* item_emb    = (const float*)d_in[1];
    const float* genre       = (const float*)d_in[2];
    const float* age         = (const float*)d_in[3];
    const float* occ         = (const float*)d_in[4];
    const float* Wg          = (const float*)d_in[5];
    const float* bg          = (const float*)d_in[6];
    const float* Wa          = (const float*)d_in[7];
    const float* ba          = (const float*)d_in[8];
    const float* Wo          = (const float*)d_in[9];
    const float* bo          = (const float*)d_in[10];
    const float* user_degree = (const float*)d_in[11];
    const float* adj_vals    = (const float*)d_in[12];
    const int*   cluster     = (const int*)d_in[13];
    const int*   adj_row     = (const int*)d_in[14];
    const int*   adj_col     = (const int*)d_in[15];
    float* out = (float*)d_out;

    const size_t N = (size_t)NUM_NODES_C * EMB;   // 19,200,000 floats
    float*    cur   = (float*)d_ws;
    float*    nxt   = cur  + N;
    float*    deg   = nxt  + N;
    float*    gsum  = deg  + NUM_USERS_C;
    float*    gcnt  = gsum + NUM_CLUSTERS_C * EMB;
    unsigned* hist  = (unsigned*)(gcnt + NUM_CLUSTERS_C + 2);
    unsigned* state = hist + 256;
    unsigned* qbits = state + 2;

    // --- quantile(user_degree, 0.75): radix-select ranks 74999 and 75000 ---
    hipMemsetAsync(hist, 0, 256 * sizeof(unsigned), stream);
    const int ranks[2] = {74999, 75000};
    for (int which = 0; which < 2; ++which) {
        select_init_kernel<<<1, 32, 0, stream>>>(state, ranks[which]);
        for (int pass = 0; pass < 4; ++pass) {
            hist_kernel<<<391, 256, 0, stream>>>(user_degree, hist, state, pass, NUM_USERS_C);
            pick_kernel<<<1, 256, 0, stream>>>(hist, state, pass, qbits, which);
        }
    }
    deg_kernel<<<(NUM_USERS_C + 255) / 256, 256, 0, stream>>>(user_degree, qbits, deg, NUM_USERS_C);

    // --- u0 = user_emb + age@Wa.T + ba + occ@Wo.T + bo  (WMMA, K padded 28->32) ---
    feat_wmma_kernel<<<NUM_USERS_C / 16, 256, 0, stream>>>(age, 7, occ, 21, Wa, Wo, ba, bo,
                                                           user_emb, cur);
    // --- cluster stats + fused blend (overwrites cur[0:NUM_USERS]) ---
    hipMemsetAsync(gsum, 0, (NUM_CLUSTERS_C * EMB + NUM_CLUSTERS_C) * sizeof(float), stream);
    cluster_kernel<<<(NUM_USERS_C + 511) / 512, 128, 0, stream>>>(cur, deg, cluster, gsum, gcnt);
    fused_kernel<<<(NUM_USERS_C * EMB + 255) / 256, 256, 0, stream>>>(cur, deg, cluster, gsum, gcnt);

    // --- i0 = item_emb + genre@Wg.T + bg  (WMMA, K padded 19->32) ---
    feat_wmma_kernel<<<NUM_ITEMS_C / 16, 256, 0, stream>>>(genre, 19, nullptr, 0, Wg, nullptr,
                                                           bg, nullptr, item_emb,
                                                           cur + (size_t)NUM_USERS_C * EMB);

    // --- acc = x; 3x { cur = A@cur; acc += cur }; out = acc/4 ---
    copy_kernel<<<(unsigned)((N + 255) / 256), 256, 0, stream>>>(out, cur, N);
    for (int layer = 0; layer < 3; ++layer) {
        hipMemsetAsync(nxt, 0, N * sizeof(float), stream);
        spmm_kernel<<<(NNZ_TOT + 7) / 8, dim3(32, 8), 0, stream>>>(adj_vals, adj_row, adj_col,
                                                                   cur, nxt, NNZ_TOT);
        if (layer < 2) {
            add_kernel<<<(unsigned)((N + 255) / 256), 256, 0, stream>>>(out, nxt, N);
            float* t = cur; cur = nxt; nxt = t;
        } else {
            add_scale_kernel<<<(unsigned)((N + 255) / 256), 256, 0, stream>>>(out, nxt, 0.25f, N);
        }
    }
}
